// VanillaMPNN_22917945491538
// MI455X (gfx1250) — matrix operand order repaired and checked by reading, not compiled, not run on hardware
//
#include <hip/hip_runtime.h>

#define N_NODES 50000
#define N_EDGES 800000
#define HIDDEN  64
#define LAYERS  4
#define IN_DIM  121
#define OUT_DIM 3

typedef __attribute__((ext_vector_type(16))) _Float16 v16h;
typedef __attribute__((ext_vector_type(8)))  _Float16 v8h;
typedef __attribute__((ext_vector_type(8)))  float    v8f;

// Fast SiLU: v * rcp(1 + exp(-v)) using hardware v_rcp_f32 / v_exp_f32.
__device__ __forceinline__ float silu(float v) {
    return v * __builtin_amdgcn_rcpf(1.0f + __expf(-v));
}

__device__ __forceinline__ void cvt8(const float4 f0, const float4 f1, v16h& a, int base) {
    a[base + 0] = (_Float16)f0.x; a[base + 1] = (_Float16)f0.y;
    a[base + 2] = (_Float16)f0.z; a[base + 3] = (_Float16)f0.w;
    a[base + 4] = (_Float16)f1.x; a[base + 5] = (_Float16)f1.y;
    a[base + 6] = (_Float16)f1.z; a[base + 7] = (_Float16)f1.w;
}

// ---------------- Embedding: x = SiLU(cat[onehot,pos] @ W + b) ----------------
__global__ void k_embed(const float* __restrict__ oh, const float* __restrict__ pos,
                        const float* __restrict__ W, const float* __restrict__ b,
                        float* __restrict__ x, float* __restrict__ sc,
                        _Float16* __restrict__ xh) {
    __shared__ float row[IN_DIM];
    const int node = blockIdx.x;
    const int t = threadIdx.x;
    for (int k = t; k < IN_DIM; k += 64)
        row[k] = (k < 118) ? oh[(size_t)node * 118 + k] : pos[(size_t)node * 3 + (k - 118)];
    __syncthreads();
    float acc = b[t];
#pragma unroll 11
    for (int k = 0; k < IN_DIM; ++k)
        acc = fmaf(row[k], W[k * HIDDEN + t], acc);
    const float v = silu(acc);
    x[(size_t)node * HIDDEN + t]  = v;
    sc[(size_t)node * HIDDEN + t] = v;
    xh[(size_t)node * HIDDEN + t] = (_Float16)v;
}

__global__ void k_zero(float* __restrict__ p, int n) {
    int i = blockIdx.x * blockDim.x + threadIdx.x;
    const int stride = gridDim.x * blockDim.x;
    for (; i < n; i += stride) p[i] = 0.0f;
}

#define WAVES_PER_BLOCK 8

// ---------------- Edge messages (hot loop, WMMA f16) ----------------
// One wave per 16-edge tile. Layer weight (128x64 f16) is register-resident
// (128 VGPRs/lane, no spills under launch_bounds(256,1)). Node features are
// gathered from the f16 mirror xh: two 16B v8h loads per k-step, shuffled
// straight into the A fragment -- no f32->f16 conversion in the hot loop.
__global__ void __launch_bounds__(256, 1)
k_edge_msg(const _Float16* __restrict__ xh,
           const int* __restrict__ ei,     // [2*E]: src row, then dst row
           const float* __restrict__ W,    // [2H, H] row-major
           const float* __restrict__ b,    // [H]
           float* __restrict__ macc) {
    __shared__ _Float16 lWt[2 * HIDDEN * HIDDEN];  // Wt[n][k]: n in [0,64), k in [0,128)
    __shared__ float    lB[HIDDEN];

    const int tid = threadIdx.x;
    for (int e = tid; e < 2 * HIDDEN * HIDDEN; e += blockDim.x) {
        const int k = e >> 6;    // feature row 0..127
        const int n = e & 63;    // output col 0..63
        lWt[n * 128 + k] = (_Float16)W[e];
    }
    if (tid < HIDDEN) lB[tid] = b[tid];
    __syncthreads();

    const int lane = tid & 31;
    const int wave = tid >> 5;
    const int mrow = lane & 15;   // M row (A-load) / N col within tile (B,C)
    const int sel  = lane >> 4;   // K-half selector

    // Preload all 16 B fragments + biases into registers (loop-invariant).
    v16h bfr[4][4];
    float biasv[4];
#pragma unroll
    for (int nt = 0; nt < 4; ++nt) {
        biasv[nt] = lB[nt * 16 + mrow];
#pragma unroll
        for (int t = 0; t < 4; ++t)
            bfr[nt][t] = *(const v16h*)&lWt[(nt * 16 + mrow) * 128 + t * 32 + sel * 16];
    }

    const int nTiles  = N_EDGES / 16;                      // 50000 exact
    const int gwave   = blockIdx.x * WAVES_PER_BLOCK + wave;
    const int wstride = gridDim.x * WAVES_PER_BLOCK;

    for (int tile = gwave; tile < nTiles; tile += wstride) {
        const int e0 = tile * 16;
        const int eA = e0 + mrow;
        const int s  = ei[eA];
        const int d  = ei[N_EDGES + eA];
        const _Float16* rs = xh + (size_t)s * HIDDEN;
        const _Float16* rd = xh + (size_t)d * HIDDEN;

        v8f c[4] = {};   // zero C (SRC2 inline 0); bias folded into epilogue

        // Stream A one k-fragment at a time: two v8h (16B) loads, no cvt.
#pragma unroll
        for (int t = 0; t < 4; ++t) {
            const _Float16* r = (t < 2) ? rs : rd;
            const int c0 = (t & 1) * 32;
            const v8h h0 = *(const v8h*)(r + c0 + sel * 16);       // K=c0+sel*16+0..7
            const v8h h1 = *(const v8h*)(r + c0 + sel * 16 + 8);   // K=c0+sel*16+8..15
            const v16h a = __builtin_shufflevector(h0, h1,
                0, 1, 2, 3, 4, 5, 6, 7, 8, 9, 10, 11, 12, 13, 14, 15);
#pragma unroll
            for (int nt = 0; nt < 4; ++nt)
                c[nt] = __builtin_amdgcn_wmma_f32_16x16x32_f16(
                            false, a, false, bfr[nt][t], (short)0, c[nt],
                            false, false);
        }

#pragma unroll
        for (int nt = 0; nt < 4; ++nt) {
            const int col = nt * 16 + mrow;
#pragma unroll
            for (int r = 0; r < 8; ++r) {
                // src id of C-row M = 8*sel + r was loaded by lane (8*sel + r).
                const int tgt = __shfl(s, sel * 8 + r, 32);
                const float v = silu(c[nt][r] + biasv[nt]);
                atomicAdd(&macc[(size_t)tgt * HIDDEN + col], v);
            }
        }
    }
}

// ---------------- Node update (WMMA f16): x += SiLU((x + 0.25*m) @ U + b) ----------------
// Also refreshes the f16 mirror xh for the next layer's edge gather.
__global__ void __launch_bounds__(256, 1)
k_update(float* __restrict__ x,
         _Float16* __restrict__ xh,
         const float* __restrict__ macc,
         const float* __restrict__ U,   // [H, H] row-major
         const float* __restrict__ b) { // [H]
    __shared__ _Float16 lUt[HIDDEN * HIDDEN];  // Ut[n][k]
    __shared__ float    lB[HIDDEN];

    const int tid = threadIdx.x;
    for (int e = tid; e < HIDDEN * HIDDEN; e += blockDim.x) {
        const int k = e >> 6;
        const int n = e & 63;
        lUt[n * 64 + k] = (_Float16)U[e];
    }
    if (tid < HIDDEN) lB[tid] = b[tid];
    __syncthreads();

    const int lane = tid & 31;
    const int wave = tid >> 5;
    const int mrow = lane & 15;
    const int sel  = lane >> 4;

    // Preload all 8 B fragments + biases into registers (loop-invariant).
    v16h bfr[4][2];
    float biasv[4];
#pragma unroll
    for (int nt = 0; nt < 4; ++nt) {
        biasv[nt] = lB[nt * 16 + mrow];
#pragma unroll
        for (int t = 0; t < 2; ++t)
            bfr[nt][t] = *(const v16h*)&lUt[(nt * 16 + mrow) * 64 + t * 32 + sel * 16];
    }

    const int nTiles  = N_NODES / 16;                      // 3125 exact
    const int gwave   = blockIdx.x * WAVES_PER_BLOCK + wave;
    const int wstride = gridDim.x * WAVES_PER_BLOCK;

    for (int tile = gwave; tile < nTiles; tile += wstride) {
        const int node0 = tile * 16;
        const float* rx = x    + (size_t)(node0 + mrow) * HIDDEN;
        const float* rm = macc + (size_t)(node0 + mrow) * HIDDEN;

        v8f c[4] = {};

#pragma unroll
        for (int t = 0; t < 2; ++t) {
            const int c0 = t * 32;
            const float4* gx0 = (const float4*)(rx + c0 + sel * 16);
            const float4* gx1 = (const float4*)(rx + c0 + sel * 16 + 8);
            const float4* gm0 = (const float4*)(rm + c0 + sel * 16);
            const float4* gm1 = (const float4*)(rm + c0 + sel * 16 + 8);
            const float4 x0 = gx0[0], x1 = gx0[1], x2 = gx1[0], x3 = gx1[1];
            const float4 m0 = gm0[0], m1 = gm0[1], m2 = gm1[0], m3 = gm1[1];
            const float4 v0 = {x0.x + 0.25f * m0.x, x0.y + 0.25f * m0.y,
                               x0.z + 0.25f * m0.z, x0.w + 0.25f * m0.w};
            const float4 v1 = {x1.x + 0.25f * m1.x, x1.y + 0.25f * m1.y,
                               x1.z + 0.25f * m1.z, x1.w + 0.25f * m1.w};
            const float4 v2 = {x2.x + 0.25f * m2.x, x2.y + 0.25f * m2.y,
                               x2.z + 0.25f * m2.z, x2.w + 0.25f * m2.w};
            const float4 v3 = {x3.x + 0.25f * m3.x, x3.y + 0.25f * m3.y,
                               x3.z + 0.25f * m3.z, x3.w + 0.25f * m3.w};
            v16h a;
            cvt8(v0, v1, a, 0);
            cvt8(v2, v3, a, 8);
#pragma unroll
            for (int nt = 0; nt < 4; ++nt)
                c[nt] = __builtin_amdgcn_wmma_f32_16x16x32_f16(
                            false, a, false, bfr[nt][t], (short)0, c[nt],
                            false, false);
        }

#pragma unroll
        for (int nt = 0; nt < 4; ++nt) {
            const int col = nt * 16 + mrow;
#pragma unroll
            for (int r = 0; r < 8; ++r) {
                const size_t idx = (size_t)(node0 + sel * 8 + r) * HIDDEN + col;
                const float nv = x[idx] + silu(c[nt][r] + biasv[nt]);  // residual
                x[idx]  = nv;
                xh[idx] = (_Float16)nv;
            }
        }
    }
}

// ---------------- Readout: out = (sc + x) @ out_W + out_b ----------------
__global__ void k_out(const float* __restrict__ x, const float* __restrict__ sc,
                      const float* __restrict__ W, const float* __restrict__ b,
                      float* __restrict__ out) {
    const int node = blockIdx.x * blockDim.x + threadIdx.x;
    if (node >= N_NODES) return;
    float a0 = b[0], a1 = b[1], a2 = b[2];
    const float* rx = x  + (size_t)node * HIDDEN;
    const float* rs = sc + (size_t)node * HIDDEN;
#pragma unroll
    for (int k = 0; k < HIDDEN; ++k) {
        const float v = rx[k] + rs[k];
        a0 = fmaf(v, W[k * OUT_DIM + 0], a0);
        a1 = fmaf(v, W[k * OUT_DIM + 1], a1);
        a2 = fmaf(v, W[k * OUT_DIM + 2], a2);
    }
    out[(size_t)node * OUT_DIM + 0] = a0;
    out[(size_t)node * OUT_DIM + 1] = a1;
    out[(size_t)node * OUT_DIM + 2] = a2;
}

extern "C" void kernel_launch(void* const* d_in, const int* in_sizes, int n_in,
                              void* d_out, int out_size, void* d_ws, size_t ws_size,
                              hipStream_t stream) {
    (void)in_sizes; (void)n_in; (void)out_size; (void)ws_size;

    const float* oh   = (const float*)d_in[0];   // [N, 118]
    const float* pos  = (const float*)d_in[1];   // [N, 3]
    const int*   ei   = (const int*)d_in[2];     // [2, E]
    const float* embW = (const float*)d_in[3];   // [121, 64]
    const float* embb = (const float*)d_in[4];   // [64]
    const float* iW   = (const float*)d_in[5];   // [L, 128, 64]
    const float* ib   = (const float*)d_in[6];   // [L, 64]
    const float* uW   = (const float*)d_in[7];   // [L, 64, 64]
    const float* ub   = (const float*)d_in[8];   // [L, 64]
    const float* oW   = (const float*)d_in[9];   // [64, 3]
    const float* ob   = (const float*)d_in[10];  // [3]
    float* out = (float*)d_out;

    float*    x   = (float*)d_ws;                        // [N, H] f32
    float*    sc  = x  + (size_t)N_NODES * HIDDEN;       // [N, H] f32
    float*    mac = sc + (size_t)N_NODES * HIDDEN;       // [N, H] f32
    _Float16* xhp = (_Float16*)(mac + (size_t)N_NODES * HIDDEN);  // [N, H] f16 mirror

    k_embed<<<N_NODES, 64, 0, stream>>>(oh, pos, embW, embb, x, sc, xhp);

    for (int l = 0; l < LAYERS; ++l) {
        k_zero<<<512, 256, 0, stream>>>(mac, N_NODES * HIDDEN);
        k_edge_msg<<<512, 256, 0, stream>>>(
            xhp, ei, iW + (size_t)l * 2 * HIDDEN * HIDDEN, ib + (size_t)l * HIDDEN, mac);
        k_update<<<200, 256, 0, stream>>>(
            x, xhp, mac, uW + (size_t)l * HIDDEN * HIDDEN, ub + (size_t)l * HIDDEN);
    }

    k_out<<<(N_NODES + 255) / 256, 256, 0, stream>>>(x, sc, oW, ob, out);
}
